// SimpleMambaSSM_88390426952200
// MI455X (gfx1250) — matrix operand order repaired and checked by reading, not compiled, not run on hardware
//
#include <hip/hip_runtime.h>
#include <hip/hip_bf16.h>

// ---------------------------------------------------------------------------
// SimpleMambaSSM for MI455X (gfx1250, wave32, WMMA)
//
// Pipeline (all in one stream):
//   0) cvt_bf16      : x, W_dt, W_B, W_C -> bf16 copies in workspace (halves
//                      GEMM operand traffic, removes per-k cvt VALU ops)
//   1) delta_gemm    : delta = softplus(x @ W_dt^T + b_dt), 4096x1024x1024,
//                      v_wmma_f32_16x16x32_bf16, 1 M-tile x 4 N-tiles / wave
//   2) bc_gemm       : Bv/Cv = x @ W_{B,C}^T + bias (N=16, one tile each)
//   3) ssm_scan      : fused sequential scan, fp32 registers, v_exp_f32,
//                      LDS-broadcast Bv/Cv chunks, prefetch of next chunk
// ---------------------------------------------------------------------------

#define BATCH   2
#define SEQ     2048
#define DMODEL  1024
#define NSTATE  16
#define MTOT    (BATCH * SEQ)      // 4096
#define KDIM    DMODEL             // 1024
#define TCH     32                 // scan timestep chunk staged in LDS

typedef __attribute__((ext_vector_type(16))) __bf16 v16bf;
typedef __attribute__((ext_vector_type(8)))  __bf16 v8bf;
typedef __attribute__((ext_vector_type(4)))  __bf16 v4bf;
typedef __attribute__((ext_vector_type(8)))  float  v8f;

// ---------------------------------------------------------------------------
// Kernel 0: fp32 -> bf16 elementwise convert (vectorized x4)
// ---------------------------------------------------------------------------
__global__ void cvt_bf16(const float* __restrict__ src,
                         __bf16* __restrict__ dst, int n4) {
    int i = blockIdx.x * blockDim.x + threadIdx.x;
    if (i < n4) {
        float4 v = ((const float4*)src)[i];
        v4bf o;
        o[0] = (__bf16)v.x; o[1] = (__bf16)v.y;
        o[2] = (__bf16)v.z; o[3] = (__bf16)v.w;
        ((v4bf*)dst)[i] = o;
    }
}

// ---- A fragment: 16x32 bf16, contiguous k order, same as B -----------------------------
// lanes 0-15  : row M = lane,    K = k0+0..15
// lanes 16-31 : row M = lane-16, K = k0+16..31
__device__ __forceinline__ v16bf load_a16(const __bf16* __restrict__ row,
                                          int k0, int khalf) {
    return *(const v16bf*)(row + k0 + khalf * 16);         // 32 B load
}

// ---- B fragment: 32x16 bf16 ------------------------------------------------
// lanes 0-15 hold K = k0+0..15 of column N = lane&15; lanes 16-31 K = +16.
// Column n of B == row n of (out,in) weight matrix: one 32 B contiguous run.
__device__ __forceinline__ v16bf load_b16(const __bf16* __restrict__ wrow,
                                          int k0, int khalf) {
    return *(const v16bf*)(wrow + k0 + khalf * 16);        // 32 B load
}

// ---------------------------------------------------------------------------
// Kernel 1: delta = softplus(x @ W_dt^T + b_dt)   (M=4096, N=1024, K=1024)
// grid = (M/16, 2), block = 256 (8 waves). Wave computes 1 M-tile x 4 N-tiles,
// reusing its A fragment across 4 WMMAs per k-step.
// ---------------------------------------------------------------------------
__global__ void delta_gemm_wmma(const __bf16* __restrict__ xb,
                                const __bf16* __restrict__ wdtb,
                                const float*  __restrict__ b_dt,
                                float* __restrict__ delta) {
    const int lane   = threadIdx.x & 31;
    const int wave   = threadIdx.x >> 5;
    const int mtile  = blockIdx.x;                       // 0..255
    const int ntbase = (blockIdx.y * 8 + wave) * 4;      // first of 4 n-tiles
    const int l15    = lane & 15;
    const int khalf  = lane >> 4;

    const __bf16* xrow = xb + (size_t)(mtile * 16 + l15) * KDIM;
    const __bf16* wrow[4];
    v8f acc[4];
#pragma unroll
    for (int j = 0; j < 4; ++j) {
        wrow[j] = wdtb + (size_t)((ntbase + j) * 16 + l15) * KDIM;
        const float bias = b_dt[(ntbase + j) * 16 + l15];
#pragma unroll
        for (int i = 0; i < 8; ++i) acc[j][i] = bias;
    }

#pragma unroll 4
    for (int k0 = 0; k0 < KDIM; k0 += 32) {
        const v16bf a = load_a16(xrow, k0, khalf);
#pragma unroll
        for (int j = 0; j < 4; ++j) {
            const v16bf b = load_b16(wrow[j], k0, khalf);
            acc[j] = __builtin_amdgcn_wmma_f32_16x16x32_bf16(
                         false, a, false, b, (short)0, acc[j], false, false);
        }
    }

    // C/D layout: VGPR i -> row = i + khalf*8, col = lane&15. Fused softplus.
#pragma unroll
    for (int j = 0; j < 4; ++j) {
#pragma unroll
        for (int i = 0; i < 8; ++i) {
            const int row = khalf * 8 + i;
            float v  = acc[j][i];
            float sp = (v > 20.f) ? v : log1pf(__expf(v));
            delta[(size_t)(mtile * 16 + row) * DMODEL + (ntbase + j) * 16 + l15] = sp;
        }
    }
}

// ---------------------------------------------------------------------------
// Kernel 2: Bv = x @ W_B^T + b_B, Cv = x @ W_C^T + b_C  (N=16, one tile each)
// grid = 32, block = 256; wave w -> m-tile blockIdx.x*8 + w. A shared by both.
// ---------------------------------------------------------------------------
__global__ void bc_gemm_wmma(const __bf16* __restrict__ xb,
                             const __bf16* __restrict__ wbb,
                             const float*  __restrict__ b_B,
                             const __bf16* __restrict__ wcb,
                             const float*  __restrict__ b_C,
                             float* __restrict__ Bv,
                             float* __restrict__ Cv) {
    const int lane  = threadIdx.x & 31;
    const int wave  = threadIdx.x >> 5;
    const int mtile = blockIdx.x * 8 + wave;             // 0..255
    const int l15   = lane & 15;
    const int khalf = lane >> 4;

    const __bf16* xrow  = xb  + (size_t)(mtile * 16 + l15) * KDIM;
    const __bf16* wbrow = wbb + (size_t)l15 * KDIM;
    const __bf16* wcrow = wcb + (size_t)l15 * KDIM;

    const float biasB = b_B[l15];
    const float biasC = b_C[l15];
    v8f cb, cc;
#pragma unroll
    for (int i = 0; i < 8; ++i) { cb[i] = biasB; cc[i] = biasC; }

#pragma unroll 4
    for (int k0 = 0; k0 < KDIM; k0 += 32) {
        const v16bf a  = load_a16(xrow, k0, khalf);
        const v16bf bb = load_b16(wbrow, k0, khalf);
        const v16bf bc = load_b16(wcrow, k0, khalf);
        cb = __builtin_amdgcn_wmma_f32_16x16x32_bf16(false, a, false, bb,
                                                     (short)0, cb, false, false);
        cc = __builtin_amdgcn_wmma_f32_16x16x32_bf16(false, a, false, bc,
                                                     (short)0, cc, false, false);
    }

#pragma unroll
    for (int i = 0; i < 8; ++i) {
        const int row = khalf * 8 + i;
        const size_t o = (size_t)(mtile * 16 + row) * NSTATE + l15;
        Bv[o] = cb[i];
        Cv[o] = cc[i];
    }
}

// ---------------------------------------------------------------------------
// Kernel 3: fused sequential scan. One thread per (b, d); h[16] in registers;
// A row hoisted; Bv/Cv chunks LDS-broadcast; next chunk prefetched.
// grid = 8, block = 256 (each block uniform in b).
// ---------------------------------------------------------------------------
__global__ void ssm_scan(const float* __restrict__ x,
                         const float* __restrict__ delta,
                         const float* __restrict__ Bv,
                         const float* __restrict__ Cv,
                         const float* __restrict__ log_A,
                         const float* __restrict__ D_skip,
                         float* __restrict__ out) {
    __shared__ float sB[TCH * NSTATE];
    __shared__ float sC[TCH * NSTATE];

    const int tid = blockIdx.x * blockDim.x + threadIdx.x;   // 0..2047
    const int b   = tid >> 10;                               // uniform per block
    const int d   = tid & (DMODEL - 1);

    float A[NSTATE];
#pragma unroll
    for (int n = 0; n < NSTATE; ++n)
        A[n] = -__expf(log_A[(size_t)d * NSTATE + n]);

    float h[NSTATE];
#pragma unroll
    for (int n = 0; n < NSTATE; ++n) h[n] = 0.f;

    const float dsk = D_skip[d];
    const float* xb = x     + (size_t)b * SEQ * DMODEL;
    const float* db = delta + (size_t)b * SEQ * DMODEL;
    const float* Bb = Bv    + (size_t)b * SEQ * NSTATE;
    const float* Cb = Cv    + (size_t)b * SEQ * NSTATE;
    float*       ob = out   + (size_t)b * SEQ * DMODEL;

    for (int t0 = 0; t0 < SEQ; t0 += TCH) {
        // Prefetch next chunk's per-thread delta/x stream (global_prefetch_b8).
        if (t0 + TCH < SEQ) {
            __builtin_prefetch(db + (size_t)(t0 + TCH) * DMODEL + d, 0, 3);
            __builtin_prefetch(xb + (size_t)(t0 + TCH) * DMODEL + d, 0, 3);
        }
        __syncthreads();
        for (int i = threadIdx.x; i < TCH * NSTATE; i += blockDim.x) {
            sB[i] = Bb[(size_t)t0 * NSTATE + i];
            sC[i] = Cb[(size_t)t0 * NSTATE + i];
        }
        __syncthreads();

        for (int tt = 0; tt < TCH; ++tt) {
            const int t = t0 + tt;
            const float dlt = db[(size_t)t * DMODEL + d];
            const float xv  = xb[(size_t)t * DMODEL + d];
            const float dbx = dlt * xv;
            float y = 0.f;
#pragma unroll
            for (int n = 0; n < NSTATE; ++n) {
                const float dA = __expf(dlt * A[n]);       // v_exp_f32
                h[n] = dA * h[n] + dbx * sB[tt * NSTATE + n];
                y    = fmaf(h[n], sC[tt * NSTATE + n], y);
            }
            ob[(size_t)t * DMODEL + d] = y + xv * dsk;
        }
    }
}

// ---------------------------------------------------------------------------
// Launch. Inputs (setup_inputs order):
//   0:x 1:W_B 2:b_B 3:W_C 4:b_C 5:W_dt 6:b_dt 7:log_A 8:D_skip
// Workspace layout (bytes):
//   delta fp32 : 4096*1024*4 = 16 MB
//   Bv, Cv     : 4096*16*4 each
//   xb   bf16  : 4096*1024*2
//   wdtb bf16  : 1024*1024*2
//   wbb, wcb   : 16*1024*2 each                (total ~27 MB)
// ---------------------------------------------------------------------------
extern "C" void kernel_launch(void* const* d_in, const int* in_sizes, int n_in,
                              void* d_out, int out_size, void* d_ws, size_t ws_size,
                              hipStream_t stream) {
    const float* x      = (const float*)d_in[0];
    const float* W_B    = (const float*)d_in[1];
    const float* b_B    = (const float*)d_in[2];
    const float* W_C    = (const float*)d_in[3];
    const float* b_C    = (const float*)d_in[4];
    const float* W_dt   = (const float*)d_in[5];
    const float* b_dt   = (const float*)d_in[6];
    const float* log_A  = (const float*)d_in[7];
    const float* D_skip = (const float*)d_in[8];
    float* out = (float*)d_out;

    float*  delta = (float*)d_ws;
    float*  Bv    = delta + (size_t)MTOT * DMODEL;
    float*  Cv    = Bv    + (size_t)MTOT * NSTATE;
    __bf16* xbf   = (__bf16*)(Cv + (size_t)MTOT * NSTATE);
    __bf16* wdtb  = xbf  + (size_t)MTOT * KDIM;
    __bf16* wbb   = wdtb + (size_t)DMODEL * KDIM;
    __bf16* wcb   = wbb  + (size_t)NSTATE * KDIM;

    // 0) fp32 -> bf16 copies
    cvt_bf16<<<dim3((MTOT * KDIM / 4 + 255) / 256), 256, 0, stream>>>(x, xbf, MTOT * KDIM / 4);
    cvt_bf16<<<dim3((DMODEL * KDIM / 4 + 255) / 256), 256, 0, stream>>>(W_dt, wdtb, DMODEL * KDIM / 4);
    cvt_bf16<<<dim3((NSTATE * KDIM / 4 + 255) / 256), 256, 0, stream>>>(W_B, wbb, NSTATE * KDIM / 4);
    cvt_bf16<<<dim3((NSTATE * KDIM / 4 + 255) / 256), 256, 0, stream>>>(W_C, wcb, NSTATE * KDIM / 4);

    // 1) delta GEMM + softplus
    delta_gemm_wmma<<<dim3(MTOT / 16, 2), 256, 0, stream>>>(xbf, wdtb, b_dt, delta);

    // 2) B/C projections
    bc_gemm_wmma<<<dim3(MTOT / 128), 256, 0, stream>>>(xbf, wbb, b_B, wcb, b_C, Bv, Cv);

    // 3) fused scan
    ssm_scan<<<dim3((BATCH * DMODEL) / 256), 256, 0, stream>>>(
        x, delta, Bv, Cv, log_A, D_skip, out);
}
